// ATT_model_70317204570660
// MI455X (gfx1250) — matrix operand order repaired and checked by reading, not compiled, not run on hardware
//
#include <hip/hip_runtime.h>
#include <math.h>

// ---------------------------------------------------------------------------
// AS-Reader (BiGRU attention-sum) for MI455X / gfx1250, wave32 + WMMA bf16.
// B=32, D=1024, Q=64, E=H=384, G=3H=1152, V=50000.
// Workspace requirement: ~300 MB (see offsets in kernel_launch).
// ---------------------------------------------------------------------------

typedef unsigned short u16;
typedef __bf16 bf16;
typedef __attribute__((ext_vector_type(16))) bf16  bf16x16;
typedef __attribute__((ext_vector_type(8)))  float f32x8;
typedef __attribute__((ext_vector_type(4)))  unsigned int u32x4;

#define NB   32
#define ND   1024
#define NQ   64
#define NE   384
#define NH   384
#define NG   1152     // 3*H
#define NHH  768      // 2*H (concat)
#define NV   50000
#define VP1  50001
#define LDSK 392      // padded LDS row stride (ushorts) for h: conflict-free b128 loads
#define EPSV 1e-12f

static __device__ __forceinline__ u16 f2bf(float x) {
  unsigned u = __float_as_uint(x);
  unsigned r = u + 0x7FFFu + ((u >> 16) & 1u);   // RNE
  return (u16)(r >> 16);
}
static __device__ __forceinline__ float bf2f(u16 h) {
  return __uint_as_float(((unsigned)h) << 16);
}
static __device__ __forceinline__ float fsig(float x) {
  return 1.f / (1.f + __expf(-x));
}

// Build a 16-element bf16 fragment from two 16-byte loads (works for global & LDS).
static __device__ __forceinline__ bf16x16 load_frag2(const u16* p0, const u16* p1) {
  union { u32x4 q[2]; bf16x16 v; } u;
  u.q[0] = *(const u32x4*)p0;
  u.q[1] = *(const u32x4*)p1;
  return u.v;
}

static __device__ __forceinline__ f32x8 wmma_bf16(bf16x16 a, bf16x16 b, f32x8 c) {
  // D = A(16x32) * B(32x16) + C, f32 accumulate
  return __builtin_amdgcn_wmma_f32_16x16x32_bf16(false, a, false, b, (short)0, c, false, false);
}

// ------------------------------ utility kernels -----------------------------

__global__ void k_f32_to_bf16(const float* __restrict__ src, u16* __restrict__ dst, int n) {
  for (int i = blockIdx.x * blockDim.x + threadIdx.x; i < n; i += gridDim.x * blockDim.x)
    dst[i] = f2bf(src[i]);
}

__global__ void k_gather_emb(const int* __restrict__ tok, const u16* __restrict__ emb,
                             u16* __restrict__ X, int nrows) {
  int n = nrows * NE;
  for (int i = blockIdx.x * blockDim.x + threadIdx.x; i < n; i += gridDim.x * blockDim.x) {
    int r = i / NE, e = i - r * NE;
    X[i] = emb[(size_t)tok[r] * NE + e];
  }
}

__global__ void k_zero_f32(float* __restrict__ p, int n) {
  for (int i = blockIdx.x * blockDim.x + threadIdx.x; i < n; i += gridDim.x * blockDim.x)
    p[i] = 0.f;
}

// ------------------------- input-projection GEMM ----------------------------
// out(MxN, bf16) = A(MxK, bf16 row-major) * Bm^T + bias, with Bm stored (NxK)
// row-major (column n of the mathematical B = contiguous row n of Bm).
// One wave computes a 16x64 strip (A-fragment reuse x4). Bias (b_ih) is fused
// here so the serial GRU scan never touches it.

__global__ __launch_bounds__(256, 2)
void k_gemm_bf16(const u16* __restrict__ A, const u16* __restrict__ Bm,
                 const float* __restrict__ bias, u16* __restrict__ out,
                 int M, int N, int K) {
  const int lane = threadIdx.x & 31;
  const int hi = lane >> 4, ml = lane & 15;
  const int wv = (blockIdx.x * blockDim.x + threadIdx.x) >> 5;
  const int nw = (gridDim.x * blockDim.x) >> 5;
  const int mt_t = M >> 4, nq_t = N >> 6;
  const f32x8 zv = {};
  for (int tidx = wv; tidx < mt_t * nq_t; tidx += nw) {
    const int mt = tidx / nq_t, nq = tidx - mt * nq_t;
    f32x8 acc[4] = {zv, zv, zv, zv};
    const u16* arow = A + ((size_t)(mt * 16 + ml)) * K;
    for (int k0 = 0; k0 < K; k0 += 32) {
      bf16x16 av = load_frag2(arow + k0 + hi * 8, arow + k0 + 16 + hi * 8);
#pragma unroll
      for (int j = 0; j < 4; ++j) {
        const u16* bp = Bm + ((size_t)(nq * 64 + j * 16 + ml)) * K + k0 + hi * 8;
        acc[j] = wmma_bf16(av, load_frag2(bp, bp + 16), acc[j]);
      }
    }
#pragma unroll
    for (int j = 0; j < 4; ++j) {
      const float bv = bias[nq * 64 + j * 16 + ml];
#pragma unroll
      for (int r = 0; r < 8; ++r)
        out[((size_t)(mt * 16 + r + hi * 8)) * N + nq * 64 + j * 16 + ml] =
            f2bf(acc[j][r] + bv);
    }
  }
}

// ------------------------------- GRU scans ----------------------------------
// 4 blocks: 0=doc-fwd 1=doc-bwd 2=query-fwd 3=query-bwd. 256 threads = 8 waves.
// Wave w owns gate columns j in [48w, 48w+48) for r/z/n simultaneously, so the
// elementwise gate math is register-local. h (32x384 bf16) lives in LDS with a
// padded stride; previous h (f32) is carried in registers. W_hh streams from
// L2 every step; next-k fragments are prefetched ahead of the WMMA consume.

__global__ __launch_bounds__(256, 1)
void k_gru_scan(const u16* __restrict__ xp_df, const u16* __restrict__ xp_db,
                const u16* __restrict__ xp_qf, const u16* __restrict__ xp_qb,
                const u16* __restrict__ whh_f, const u16* __restrict__ whh_b,
                const float* __restrict__ bhh_f, const float* __restrict__ bhh_b,
                const int* __restrict__ dlens, const int* __restrict__ qlens,
                u16* __restrict__ hcat_d, u16* __restrict__ hcat_q) {
  const int dir = blockIdx.x;
  const bool isdoc = dir < 2;
  const bool bwd = (dir & 1) != 0;
  const int T = isdoc ? ND : NQ;
  const u16* xp = (dir == 0) ? xp_df : (dir == 1) ? xp_db : (dir == 2) ? xp_qf : xp_qb;
  const u16* whh = bwd ? whh_b : whh_f;
  const float* bhh = bwd ? bhh_b : bhh_f;
  const int* lens = isdoc ? dlens : qlens;
  u16* out = isdoc ? hcat_d : hcat_q;
  const int dcol = bwd ? NH : 0;

  __shared__ u16 hb[32 * LDSK];
  const int tid = threadIdx.x;
  const int lane = tid & 31, wv = tid >> 5;
  const int hi = lane >> 4, ml = lane & 15;

  for (int i = tid; i < 32 * LDSK; i += 256) hb[i] = 0;
  __syncthreads();

  float hold[2][3][8];
#pragma unroll
  for (int mt = 0; mt < 2; ++mt)
#pragma unroll
    for (int c = 0; c < 3; ++c)
#pragma unroll
      for (int r = 0; r < 8; ++r) hold[mt][c][r] = 0.f;

  const f32x8 zv = {};
  for (int t = 0; t < T; ++t) {
    f32x8 acc[2][3][3];
#pragma unroll
    for (int mt = 0; mt < 2; ++mt)
#pragma unroll
      for (int g = 0; g < 3; ++g)
#pragma unroll
        for (int c = 0; c < 3; ++c) acc[mt][g][c] = zv;

    // gh = h @ W_hh^T : (32x384)x(384x1152), wave covers its 144 columns
    for (int ks = 0; ks < 12; ++ks) {
      const int k0 = ks * 32;
      const u16* ar0 = &hb[(0 * 16 + ml) * LDSK + k0];
      const u16* ar1 = &hb[(16 + ml) * LDSK + k0];
      bf16x16 a0 = load_frag2(ar0 + hi * 8, ar0 + 16 + hi * 8);
      bf16x16 a1 = load_frag2(ar1 + hi * 8, ar1 + 16 + hi * 8);
#pragma unroll
      for (int g = 0; g < 3; ++g) {
#pragma unroll
        for (int c = 0; c < 3; ++c) {
          const u16* bp = whh + ((size_t)(g * NH + wv * 48 + c * 16 + ml)) * NH + k0 + hi * 8;
          if (ks < 11) __builtin_prefetch(bp + 32, 0, 1);  // next k-step (global_prefetch_b8)
          bf16x16 bfr = load_frag2(bp, bp + 16);
          acc[0][g][c] = wmma_bf16(a0, bfr, acc[0][g][c]);
          acc[1][g][c] = wmma_bf16(a1, bfr, acc[1][g][c]);
        }
      }
    }
    __syncthreads();  // all A-fragment reads done before h is overwritten

#pragma unroll
    for (int mt = 0; mt < 2; ++mt) {
#pragma unroll
      for (int c = 0; c < 3; ++c) {
#pragma unroll
        for (int r = 0; r < 8; ++r) {
          const int bb = mt * 16 + r + hi * 8;          // batch row
          const int j = wv * 48 + c * 16 + ml;          // hidden column
          const int len = lens[bb];
          const int trow = bwd ? ((t < len) ? (len - 1 - t) : t) : t;
          const size_t xb = ((size_t)bb * T + trow) * NG;
          float ir = bf2f(xp[xb + j]);                  // b_ih already fused into xp
          float iz = bf2f(xp[xb + NH + j]);
          float in_ = bf2f(xp[xb + 2 * NH + j]);
          float hr = acc[mt][0][c][r] + bhh[j];
          float hz = acc[mt][1][c][r] + bhh[NH + j];
          float hn = acc[mt][2][c][r] + bhh[2 * NH + j];
          float rg = fsig(ir + hr);
          float zg = fsig(iz + hz);
          float ng = tanhf(in_ + rg * hn);
          float hv = (1.f - zg) * ng + zg * hold[mt][c][r];
          hold[mt][c][r] = hv;
          hb[bb * LDSK + j] = f2bf(hv);
          const int orow = bwd ? trow : t;               // bwd writes to reversed pos
          float ov = (t < len) ? hv : 0.f;               // valid-mask the output
          out[((size_t)bb * T + orow) * NHH + dcol + j] = f2bf(ov);
        }
      }
    }
    __syncthreads();
  }
}

// ------------------------------ scores GEMM ---------------------------------
// scores[b,d,q] = sum_h hcat_d[b,d,h]*hcat_q[b,q,h], K=768. Per-batch tiles:
// grid = 32*8 blocks of 8 waves; wave = one 16x64 M-strip.

__global__ __launch_bounds__(256, 2)
void k_scores(const u16* __restrict__ hd, const u16* __restrict__ hq,
              float* __restrict__ scores) {
  const int b = blockIdx.x >> 3;
  const int mt = ((blockIdx.x & 7) << 3) + (threadIdx.x >> 5);  // 0..63
  const int lane = threadIdx.x & 31, hi = lane >> 4, ml = lane & 15;
  const u16* A = hd + (size_t)b * ND * NHH;
  const u16* Bm = hq + (size_t)b * NQ * NHH;
  const f32x8 zv = {};
  f32x8 acc[4] = {zv, zv, zv, zv};
  const u16* arow = A + ((size_t)(mt * 16 + ml)) * NHH;
  for (int k0 = 0; k0 < NHH; k0 += 32) {
    bf16x16 av = load_frag2(arow + k0 + hi * 8, arow + k0 + 16 + hi * 8);
#pragma unroll
    for (int j = 0; j < 4; ++j) {
      const u16* bp = Bm + ((size_t)(j * 16 + ml)) * NHH + k0 + hi * 8;
      acc[j] = wmma_bf16(av, load_frag2(bp, bp + 16), acc[j]);
    }
  }
  float* so = scores + (size_t)b * ND * NQ;
#pragma unroll
  for (int j = 0; j < 4; ++j)
#pragma unroll
    for (int r = 0; r < 8; ++r)
      so[((size_t)(mt * 16 + r + hi * 8)) * NQ + j * 16 + ml] = acc[j][r];
}

// ------------------- softmaxes / beta_aver / s / probs ----------------------
// One block (256 threads) per batch.

__global__ __launch_bounds__(256, 1)
void k_combine(const float* __restrict__ scores, const float* __restrict__ dmask,
               const float* __restrict__ qmask, const int* __restrict__ documents,
               const int* __restrict__ answers, const int* __restrict__ dlens,
               float* __restrict__ s_out, float* __restrict__ probs_out) {
  const int b = blockIdx.x, tid = threadIdx.x;
  const float* sc = scores + (size_t)b * ND * NQ;
  const float* dm = dmask + (size_t)b * ND;
  const float* qm = qmask + (size_t)b * NQ;
  __shared__ float cmax[NQ], csum[NQ], ba[NQ];
  __shared__ float rmax[ND], rsum[ND];
  __shared__ float red[256];

  // column stats (alpha: softmax over d); shift is the UNMASKED max (per ref)
  if (tid < NQ) {
    float mx = -3.0e38f;
    for (int d = 0; d < ND; ++d) mx = fmaxf(mx, sc[d * NQ + tid]);
    float sm = 0.f;
    const float qv = qm[tid];
    for (int d = 0; d < ND; ++d) sm += __expf(sc[d * NQ + tid] - mx) * dm[d] * qv;
    cmax[tid] = mx;
    csum[tid] = sm;
  }
  // row stats (beta: softmax over q)
  for (int d = tid; d < ND; d += 256) {
    float mx = -3.0e38f;
    for (int q = 0; q < NQ; ++q) mx = fmaxf(mx, sc[d * NQ + q]);
    float sm = 0.f;
    const float dv = dm[d];
    for (int q = 0; q < NQ; ++q) sm += __expf(sc[d * NQ + q] - mx) * dv * qm[q];
    rmax[d] = mx;
    rsum[d] = sm;
  }
  __syncthreads();
  // beta_aver[q] = (1/len) * sum_d beta[d,q]
  if (tid < NQ) {
    float acc = 0.f;
    const float qv = qm[tid];
    for (int d = 0; d < ND; ++d)
      acc += __expf(sc[d * NQ + tid] - rmax[d]) * dm[d] * qv / (rsum[d] + EPSV);
    ba[tid] = acc / (float)dlens[b];
  }
  __syncthreads();
  // s[d] = sum_q alpha[d,q] * beta_aver[q]; probs = sum_d s[d]*(doc==ans)
  float pr = 0.f;
  const int ans = answers[b];
  for (int d = tid; d < ND; d += 256) {
    float sd = 0.f;
    const float dv = dm[d];
    for (int q = 0; q < NQ; ++q)
      sd += __expf(sc[d * NQ + q] - cmax[q]) * dv * qm[q] / (csum[q] + EPSV) * ba[q];
    s_out[(size_t)b * ND + d] = sd;
    pr += sd * ((documents[(size_t)b * ND + d] == ans) ? 1.f : 0.f);
  }
  red[tid] = pr;
  __syncthreads();
  for (int s = 128; s > 0; s >>= 1) {
    if (tid < s) red[tid] += red[tid + s];
    __syncthreads();
  }
  if (tid == 0) probs_out[b] = red[0];
}

// ------------------------------ pick (argmax) -------------------------------

__global__ void k_scatter(const int* __restrict__ documents, const float* __restrict__ dmask,
                          const float* __restrict__ s_vec, float* __restrict__ scb,
                          float* __restrict__ cntb) {
  const int n = NB * ND;
  for (int i = blockIdx.x * blockDim.x + threadIdx.x; i < n; i += gridDim.x * blockDim.x) {
    float v = dmask[i];
    if (v > 0.f) {
      int bb = i >> 10;
      int tok = documents[i];
      atomicAdd(&scb[(size_t)bb * VP1 + tok], s_vec[i] * v);
      atomicAdd(&cntb[(size_t)bb * VP1 + tok], v);
    }
  }
}

__global__ __launch_bounds__(256, 1)
void k_argmax(const float* __restrict__ scb, const float* __restrict__ cntb,
              float* __restrict__ outp) {
  const int b = blockIdx.x, tid = threadIdx.x;
  const float* sc = scb + (size_t)b * VP1;
  const float* cn = cntb + (size_t)b * VP1;
  float bv = -3.0e38f;
  int bi = 0x7fffffff;
  for (int i = tid; i < VP1; i += 256) {
    float val = (cn[i] > 0.f) ? sc[i] : -3.0e38f;
    if (val > bv || (val == bv && i < bi)) { bv = val; bi = i; }
  }
  __shared__ float rv[256];
  __shared__ int ri[256];
  rv[tid] = bv; ri[tid] = bi;
  __syncthreads();
  for (int s = 128; s > 0; s >>= 1) {
    if (tid < s) {
      if (rv[tid + s] > rv[tid] || (rv[tid + s] == rv[tid] && ri[tid + s] < ri[tid])) {
        rv[tid] = rv[tid + s]; ri[tid] = ri[tid + s];
      }
    }
    __syncthreads();
  }
  if (tid == 0) outp[b] = (float)((ri[0] == 0x7fffffff) ? 0 : ri[0]);
}

// --------------------------------- launcher ---------------------------------

extern "C" void kernel_launch(void* const* d_in, const int* in_sizes, int n_in,
                              void* d_out, int out_size, void* d_ws, size_t ws_size,
                              hipStream_t stream) {
  (void)in_sizes; (void)n_in; (void)out_size; (void)ws_size;
  const int*   documents  = (const int*)d_in[0];
  const int*   doc_lens   = (const int*)d_in[1];
  const float* doc_masks  = (const float*)d_in[2];
  const int*   querys     = (const int*)d_in[3];
  const int*   query_lens = (const int*)d_in[4];
  const float* query_masks= (const float*)d_in[5];
  const int*   answers    = (const int*)d_in[6];
  const float* emb        = (const float*)d_in[7];
  const float* w_ih_f     = (const float*)d_in[8];
  const float* w_hh_f     = (const float*)d_in[9];
  const float* b_ih_f     = (const float*)d_in[10];
  const float* b_hh_f     = (const float*)d_in[11];
  const float* w_ih_b     = (const float*)d_in[12];
  const float* w_hh_b     = (const float*)d_in[13];
  const float* b_ih_b     = (const float*)d_in[14];
  const float* b_hh_b     = (const float*)d_in[15];

  // ---- workspace carve-out (~300 MB total) ----
  char* ws = (char*)d_ws;
  size_t off = 0;
  auto carve = [&](size_t bytes) -> char* {
    char* p = ws + off;
    off = (off + bytes + 255) & ~(size_t)255;
    return p;
  };
  u16* emb_bf  = (u16*)carve((size_t)VP1 * NE * 2);            // 38.4 MB
  u16* Xd      = (u16*)carve((size_t)NB * ND * NE * 2);        // 25.2 MB
  u16* Xq      = (u16*)carve((size_t)NB * NQ * NE * 2);        //  1.6 MB
  u16* wihf_bf = (u16*)carve((size_t)NG * NE * 2);
  u16* wihb_bf = (u16*)carve((size_t)NG * NE * 2);
  u16* whhf_bf = (u16*)carve((size_t)NG * NH * 2);
  u16* whhb_bf = (u16*)carve((size_t)NG * NH * 2);
  u16* xp_df   = (u16*)carve((size_t)NB * ND * NG * 2);        // 75.5 MB
  u16* xp_db   = (u16*)carve((size_t)NB * ND * NG * 2);        // 75.5 MB
  u16* xp_qf   = (u16*)carve((size_t)NB * NQ * NG * 2);        //  4.7 MB
  u16* xp_qb   = (u16*)carve((size_t)NB * NQ * NG * 2);        //  4.7 MB
  u16* hcat_d  = (u16*)carve((size_t)NB * ND * NHH * 2);       // 50.3 MB
  u16* hcat_q  = (u16*)carve((size_t)NB * NQ * NHH * 2);       //  3.1 MB
  float* scores= (float*)carve((size_t)NB * ND * NQ * 4);      //  8.4 MB
  float* s_vec = (float*)carve((size_t)NB * ND * 4);
  float* scb   = (float*)carve((size_t)NB * VP1 * 4);          //  6.4 MB
  float* cntb  = (float*)carve((size_t)NB * VP1 * 4);          //  6.4 MB

  float* probs_out = (float*)d_out;        // [0:32)
  float* pred_out  = (float*)d_out + NB;   // [32:64)

  // 1) precision conversion (f32 -> bf16)
  k_f32_to_bf16<<<4096, 256, 0, stream>>>(emb, emb_bf, VP1 * NE);
  k_f32_to_bf16<<<512, 256, 0, stream>>>(w_ih_f, wihf_bf, NG * NE);
  k_f32_to_bf16<<<512, 256, 0, stream>>>(w_ih_b, wihb_bf, NG * NE);
  k_f32_to_bf16<<<512, 256, 0, stream>>>(w_hh_f, whhf_bf, NG * NH);
  k_f32_to_bf16<<<512, 256, 0, stream>>>(w_hh_b, whhb_bf, NG * NH);

  // 2) embedding gather
  k_gather_emb<<<4096, 256, 0, stream>>>(documents, emb_bf, Xd, NB * ND);
  k_gather_emb<<<512, 256, 0, stream>>>(querys, emb_bf, Xq, NB * NQ);

  // 3) input projections xp = X @ W_ih^T + b_ih (bf16 WMMA, f32 accum)
  k_gemm_bf16<<<2048, 256, 0, stream>>>(Xd, wihf_bf, b_ih_f, xp_df, NB * ND, NG, NE);
  k_gemm_bf16<<<2048, 256, 0, stream>>>(Xd, wihb_bf, b_ih_b, xp_db, NB * ND, NG, NE);
  k_gemm_bf16<<<256, 256, 0, stream>>>(Xq, wihf_bf, b_ih_f, xp_qf, NB * NQ, NG, NE);
  k_gemm_bf16<<<256, 256, 0, stream>>>(Xq, wihb_bf, b_ih_b, xp_qb, NB * NQ, NG, NE);

  // 4) four concurrent GRU scans (doc/query x fwd/bwd), WMMA per step
  k_gru_scan<<<4, 256, 0, stream>>>(xp_df, xp_db, xp_qf, xp_qb,
                                    whhf_bf, whhb_bf,
                                    b_hh_f, b_hh_b,
                                    doc_lens, query_lens, hcat_d, hcat_q);

  // 5) scores = doc_h . q_h (K=768 bf16 WMMA)
  k_scores<<<NB * 8, 256, 0, stream>>>(hcat_d, hcat_q, scores);

  // 6) dual masked softmax, beta_aver, s, probs
  k_combine<<<NB, 256, 0, stream>>>(scores, doc_masks, query_masks, documents,
                                    answers, doc_lens, s_vec, probs_out);

  // 7) attention-sum pick
  k_zero_f32<<<4096, 256, 0, stream>>>(scb, NB * VP1);
  k_zero_f32<<<4096, 256, 0, stream>>>(cntb, NB * VP1);
  k_scatter<<<512, 256, 0, stream>>>(documents, doc_masks, s_vec, scb, cntb);
  k_argmax<<<NB, 256, 0, stream>>>(scb, cntb, pred_out);
}
